// CrossFeatureAttention_2216203125226
// MI455X (gfx1250) — matrix operand order repaired and checked by reading, not compiled, not run on hardware
//
#include <hip/hip_runtime.h>

// ---------------------------------------------------------------------------
// CrossFeatureAttention for MI455X (gfx1250, wave32, WMMA 16x16x32 f16->f32)
//   B=4, N1=N2=4096, C=512
//   Q = x1*Wq^T+bq ; K = x2*Wk^T+bk ; V = x2*Wv^T+bv
//   S = softmax(Q K^T / sqrt(C)) ; out = (Q + S V) Wo^T + bo
// ---------------------------------------------------------------------------

typedef __attribute__((ext_vector_type(16))) _Float16     v16h;
typedef __attribute__((ext_vector_type(8)))  float        v8f;
typedef __attribute__((ext_vector_type(4)))  unsigned int u32x4;
typedef __attribute__((ext_vector_type(4)))  int          i32x4;
typedef __attribute__((ext_vector_type(4)))  float        f32x4;
typedef _Float16 h16;

#define AS1 __attribute__((address_space(1)))
#define AS3 __attribute__((address_space(3)))

#if __has_builtin(__builtin_amdgcn_global_load_async_to_lds_b128) && \
    __has_builtin(__builtin_amdgcn_s_wait_asynccnt)
#define HAS_ASYNC_LDS 1
#else
#define HAS_ASYNC_LDS 0
#endif

// Copy 16B global -> LDS. Uses gfx1250 async DMA (ASYNCcnt) when available.
__device__ __forceinline__ void cp16B(const h16* __restrict__ g, h16* l)
{
#if HAS_ASYNC_LDS
    __builtin_amdgcn_global_load_async_to_lds_b128(
        (AS1 i32x4*)(h16*)g, (AS3 i32x4*)l, 0, 0);
#else
    *(u32x4*)l = *(const u32x4*)g;
#endif
}

__device__ __forceinline__ void cp_wait()
{
#if HAS_ASYNC_LDS
    __builtin_amdgcn_s_wait_asynccnt(0);
#endif
}

// ===========================================================================
// Generic tensor-core GEMM:  OUT[m,n] = scale * sum_k A[m,k]*W[n,k]
//                                        (+ bias[n]) (+ resid[m,n])
//   A: M x K (row major, f32 or f16)   W: N x K (row major, f32 or f16)
//   OUT: M x N (f16 or f32).  M%128==0, N%64==0, K%64==0.
// Block: 256 threads = 8 waves; block tile 128x64, k-step 64.
// Wave (wr=wave>>1, wc=wave&1) owns a 32x32 tile -> 4 accumulators,
// 8 WMMAs per staging round (2 k-substeps of 32).
// ===========================================================================
template <int A_F32, int W_F32, int OUT_F32>
__global__ __launch_bounds__(256)
void gemm_awT(const void* __restrict__ Ap, const void* __restrict__ Wp,
              const float* __restrict__ bias, const h16* __restrict__ resid,
              void* __restrict__ outp, int M, int N, int K, float scale)
{
    constexpr int LDT = 72;               // padded row stride (halves):
                                          //  144B rows (16B aligned),
                                          //  36-dword bank stride -> 16
                                          //  consecutive rows hit 16 banks
    __shared__ __align__(16) h16 As[128 * LDT];
    __shared__ __align__(16) h16 Bs[64 * LDT];

    const int t    = threadIdx.x;
    const int lane = t & 31;
    const int wave = t >> 5;
    const int hi   = lane >> 4;           // lane group (0: lanes0-15, 1: 16-31)
    const int ln   = lane & 15;
    const int wr   = wave >> 1;           // 0..3  -> 32-row strip
    const int wc   = wave & 1;            // 0..1  -> 32-col half
    const int m0   = blockIdx.x * 128;
    const int n0   = blockIdx.y * 64;

    // staging maps
    const int arow = t >> 1;              // 0..127
    const int ako  = (t & 1) * 32;        // 0,32 : 32 halves (4x16B chunks)
    const int wrow = t >> 2;              // 0..63
    const int wko  = (t & 3) * 16;        // 0,16,32,48 : 16 halves (2 chunks)

    v8f acc00 = {}, acc01 = {}, acc10 = {}, acc11 = {};

    for (int k0 = 0; k0 < K; k0 += 64) {
        // ---- stage A tile (128 x 64 halves) ----
        if (A_F32) {
            const float* A = (const float*)Ap;
            const float* g = A + (size_t)(m0 + arow) * K + k0 + ako;
#pragma unroll
            for (int c = 0; c < 4; ++c) {
                f32x4 f0 = ((const f32x4*)(g + c * 8))[0];
                f32x4 f1 = ((const f32x4*)(g + c * 8))[1];
                union { h16 h[8]; u32x4 q; } u;
                u.h[0] = (h16)f0.x; u.h[1] = (h16)f0.y;
                u.h[2] = (h16)f0.z; u.h[3] = (h16)f0.w;
                u.h[4] = (h16)f1.x; u.h[5] = (h16)f1.y;
                u.h[6] = (h16)f1.z; u.h[7] = (h16)f1.w;
                *(u32x4*)&As[arow * LDT + ako + c * 8] = u.q;
            }
            if (k0 + 64 < K)
                __builtin_prefetch(g + 64, 0, 0);
        } else {
            const h16* A = (const h16*)Ap;
            const h16* g = A + (size_t)(m0 + arow) * K + k0 + ako;
#pragma unroll
            for (int c = 0; c < 4; ++c)
                cp16B(g + c * 8, &As[arow * LDT + ako + c * 8]);
        }
        // ---- stage W tile (64 weight rows x 64 halves) ----
        if (W_F32) {
            const float* W = (const float*)Wp;
            const float* g = W + (size_t)(n0 + wrow) * K + k0 + wko;
#pragma unroll
            for (int c = 0; c < 2; ++c) {
                f32x4 f0 = ((const f32x4*)(g + c * 8))[0];
                f32x4 f1 = ((const f32x4*)(g + c * 8))[1];
                union { h16 h[8]; u32x4 q; } u;
                u.h[0] = (h16)f0.x; u.h[1] = (h16)f0.y;
                u.h[2] = (h16)f0.z; u.h[3] = (h16)f0.w;
                u.h[4] = (h16)f1.x; u.h[5] = (h16)f1.y;
                u.h[6] = (h16)f1.z; u.h[7] = (h16)f1.w;
                *(u32x4*)&Bs[wrow * LDT + wko + c * 8] = u.q;
            }
            if (k0 + 64 < K)
                __builtin_prefetch(g + 64, 0, 0);
        } else {
            const h16* W = (const h16*)Wp;
            const h16* g = W + (size_t)(n0 + wrow) * K + k0 + wko;
#pragma unroll
            for (int c = 0; c < 2; ++c)
                cp16B(g + c * 8, &Bs[wrow * LDT + wko + c * 8]);
        }
        cp_wait();
        __syncthreads();

        // ---- compute: 2 k-substeps x 4 WMMA ----
#pragma unroll
        for (int ks = 0; ks < 2; ++ks) {
            const int kb = ks * 32;
            // A 16x32 f16 frag: lane m = ln; elem e <-> k = kb+(e&7)+16*(e>>3)+8*hi
            union { v16h v; u32x4 q[2]; } a0, a1, b0, b1;
            const int ar0 = (wr * 32 + ln) * LDT + kb + 8 * hi;
            const int ar1 = ar0 + 16 * LDT;
            a0.q[0] = *(const u32x4*)&As[ar0];
            a0.q[1] = *(const u32x4*)&As[ar0 + 16];
            a1.q[0] = *(const u32x4*)&As[ar1];
            a1.q[1] = *(const u32x4*)&As[ar1 + 16];
            // B 32x16 f16 frag: lane n = ln; elem e <-> k = kb+(e&7)+16*(e>>3)+8*hi
            const int br0 = (wc * 32 + ln) * LDT + kb + 8 * hi;
            const int br1 = br0 + 16 * LDT;
            b0.q[0] = *(const u32x4*)&Bs[br0];
            b0.q[1] = *(const u32x4*)&Bs[br0 + 16];
            b1.q[0] = *(const u32x4*)&Bs[br1];
            b1.q[1] = *(const u32x4*)&Bs[br1 + 16];

            acc00 = __builtin_amdgcn_wmma_f32_16x16x32_f16(
                false, a0.v, false, b0.v, (short)0, acc00, false, false);
            acc01 = __builtin_amdgcn_wmma_f32_16x16x32_f16(
                false, a0.v, false, b1.v, (short)0, acc01, false, false);
            acc10 = __builtin_amdgcn_wmma_f32_16x16x32_f16(
                false, a1.v, false, b0.v, (short)0, acc10, false, false);
            acc11 = __builtin_amdgcn_wmma_f32_16x16x32_f16(
                false, a1.v, false, b1.v, (short)0, acc11, false, false);
        }
        __syncthreads();
    }

    // ---- epilogue: C/D layout: VGPR v -> row m = v + 8*hi, col n = ln ----
    const int nb0 = n0 + wc * 32 + ln;
    const int nb1 = nb0 + 16;
    const float bia0 = bias ? bias[nb0] : 0.0f;
    const float bia1 = bias ? bias[nb1] : 0.0f;
#pragma unroll
    for (int ms = 0; ms < 2; ++ms) {
        const v8f& c0 = ms ? acc10 : acc00;
        const v8f& c1 = ms ? acc11 : acc01;
#pragma unroll
        for (int v = 0; v < 8; ++v) {
            const int m = m0 + wr * 32 + ms * 16 + v + 8 * hi;
            float o0 = c0[v] * scale + bia0;
            float o1 = c1[v] * scale + bia1;
            if (resid) {
                o0 += (float)resid[(size_t)m * N + nb0];
                o1 += (float)resid[(size_t)m * N + nb1];
            }
            if (OUT_F32) {
                ((float*)outp)[(size_t)m * N + nb0] = o0;
                ((float*)outp)[(size_t)m * N + nb1] = o1;
            } else {
                ((h16*)outp)[(size_t)m * N + nb0] = (h16)o0;
                ((h16*)outp)[(size_t)m * N + nb1] = (h16)o1;
            }
        }
    }
}

// ===========================================================================
// Row-wise softmax in place over f16 rows of length 4096 (one block per row).
// ===========================================================================
__global__ __launch_bounds__(256)
void softmax_rows(h16* __restrict__ S, int ncols)
{
    const int row = blockIdx.x;
    h16* p = S + (size_t)row * ncols;
    const int t = threadIdx.x;

    float vals[16];
    float mx = -3.0e38f;
#pragma unroll
    for (int j = 0; j < 16; ++j) {
        vals[j] = (float)p[t + j * 256];
        mx = fmaxf(mx, vals[j]);
    }
#pragma unroll
    for (int off = 16; off > 0; off >>= 1)
        mx = fmaxf(mx, __shfl_xor(mx, off, 32));

    __shared__ float redm[8];
    __shared__ float reds[8];
    const int wave = t >> 5, lane = t & 31;
    if (lane == 0) redm[wave] = mx;
    __syncthreads();
    mx = redm[0];
#pragma unroll
    for (int w = 1; w < 8; ++w) mx = fmaxf(mx, redm[w]);

    float sum = 0.0f;
#pragma unroll
    for (int j = 0; j < 16; ++j) {
        vals[j] = __expf(vals[j] - mx);
        sum += vals[j];
    }
#pragma unroll
    for (int off = 16; off > 0; off >>= 1)
        sum += __shfl_xor(sum, off, 32);
    if (lane == 0) reds[wave] = sum;
    __syncthreads();
    sum = 0.0f;
#pragma unroll
    for (int w = 0; w < 8; ++w) sum += reds[w];

    const float inv = 1.0f / sum;
#pragma unroll
    for (int j = 0; j < 16; ++j)
        p[t + j * 256] = (h16)(vals[j] * inv);
}

// ===========================================================================
// Per-batch transpose: Vh[b][m][d] (4096x512) -> Vt[b][d][m] (512x4096)
// so that P*V can reuse the A*W^T fragment path (rows contiguous along k).
// ===========================================================================
__global__ __launch_bounds__(256)
void transpose_kd(const h16* __restrict__ src, h16* __restrict__ dst)
{
    const size_t base = (size_t)blockIdx.z * 4096 * 512;
    const int idx = blockIdx.x * 256 + threadIdx.x;   // coalesced read
    const int m = idx >> 9;
    const int d = idx & 511;
    dst[base + (size_t)d * 4096 + m] = src[base + idx];
}

// ===========================================================================
// Host-side launcher
// ===========================================================================
extern "C" void kernel_launch(void* const* d_in, const int* in_sizes, int n_in,
                              void* d_out, int out_size, void* d_ws, size_t ws_size,
                              hipStream_t stream)
{
    (void)in_sizes; (void)n_in; (void)out_size; (void)ws_size;

    const float* x1 = (const float*)d_in[0];
    const float* x2 = (const float*)d_in[1];
    const float* Wq = (const float*)d_in[2];
    const float* bq = (const float*)d_in[3];
    const float* Wk = (const float*)d_in[4];
    const float* bk = (const float*)d_in[5];
    const float* Wv = (const float*)d_in[6];
    const float* bv = (const float*)d_in[7];
    const float* Wo = (const float*)d_in[8];
    const float* bo = (const float*)d_in[9];
    float* out = (float*)d_out;

    const int B = 4, N1 = 4096, N2 = 4096, C = 512;
    const size_t BNC = (size_t)B * N1 * C;           // 8,388,608

    // workspace layout (f16): Qh | Kh | Vh | Vt | O1 | S(per-batch 4096x4096)
    h16* Qh = (h16*)d_ws;
    h16* Kh = Qh + BNC;
    h16* Vh = Kh + BNC;
    h16* Vt = Vh + BNC;
    h16* O1 = Vt + BNC;
    h16* S  = O1 + BNC;                              // 16,777,216 halves

    const dim3 blk(256);
    const float scale = 0.044194173824159216f;       // 1/sqrt(512)

    // 1) projections: f32 inputs/weights converted to f16 while staging
    gemm_awT<1, 1, 0><<<dim3((B * N1) / 128, C / 64), blk, 0, stream>>>(
        x1, Wq, bq, nullptr, Qh, B * N1, C, C, 1.0f);
    gemm_awT<1, 1, 0><<<dim3((B * N2) / 128, C / 64), blk, 0, stream>>>(
        x2, Wk, bk, nullptr, Kh, B * N2, C, C, 1.0f);
    gemm_awT<1, 1, 0><<<dim3((B * N2) / 128, C / 64), blk, 0, stream>>>(
        x2, Wv, bv, nullptr, Vh, B * N2, C, C, 1.0f);

    // 2) transpose V per batch for the P*V GEMM
    transpose_kd<<<dim3((N2 * C) / 256, 1, B), blk, 0, stream>>>(Vh, Vt);

    // 3) per-batch attention: scores -> softmax -> P*V (+ residual Q)
    for (int b = 0; b < B; ++b) {
        const h16* Qb  = Qh + (size_t)b * N1 * C;
        const h16* Kb  = Kh + (size_t)b * N2 * C;
        const h16* Vtb = Vt + (size_t)b * N2 * C;
        h16*       O1b = O1 + (size_t)b * N1 * C;

        gemm_awT<0, 0, 0><<<dim3(N1 / 128, N2 / 64), blk, 0, stream>>>(
            Qb, Kb, nullptr, nullptr, S, N1, N2, C, scale);
        softmax_rows<<<dim3(N1), blk, 0, stream>>>(S, N2);
        gemm_awT<0, 0, 0><<<dim3(N1 / 128, C / 64), blk, 0, stream>>>(
            S, Vtb, nullptr, Qb, O1b, N1, C, N2, 1.0f);
    }

    // 4) output projection: f16 activations, f32 weights, f32 output + bias
    gemm_awT<0, 1, 1><<<dim3((B * N1) / 128, C / 64), blk, 0, stream>>>(
        O1, Wo, bo, nullptr, out, B * N1, C, C, 1.0f);
}
